// DiffGNN_8383776162491
// MI455X (gfx1250) — compile-verified, matrix operand order checked by reading, not run on hardware
//
#include <hip/hip_runtime.h>
#include <hip/hip_bf16.h>
#include <stdint.h>

typedef __attribute__((ext_vector_type(16))) __bf16        v16bf;
typedef __attribute__((ext_vector_type(8)))  float         v8f;
typedef __attribute__((ext_vector_type(4)))  unsigned int  u32x4;

static __device__ __forceinline__ unsigned short f2bf(float f) {
  // round-to-nearest-even fp32 -> bf16
  unsigned u = __builtin_bit_cast(unsigned, f);
  unsigned r = 0x7FFFu + ((u >> 16) & 1u);
  return (unsigned short)((u + r) >> 16);
}

union FragBF { v16bf v; u32x4 q[2]; };

// ---------------- prep kernels ----------------
__global__ void k_cast_bf16(const float* __restrict__ in, unsigned short* __restrict__ out, int n) {
  int i = blockIdx.x * blockDim.x + threadIdx.x;
  if (i < n) out[i] = f2bf(in[i]);
}

// W[K][Nout] fp32 row-major  ->  Wt[Nout][K] bf16 (row n contiguous in K)
__global__ void k_transpose_bf16(const float* __restrict__ W, unsigned short* __restrict__ Wt,
                                 int K, int Nout) {
  int i = blockIdx.x * blockDim.x + threadIdx.x;
  if (i >= K * Nout) return;
  int n = i / K, k = i - n * K;
  Wt[n * K + k] = f2bf(W[k * Nout + n]);
}

__global__ void k_deg_init(float* __restrict__ deg, int N) {
  int i = blockIdx.x * blockDim.x + threadIdx.x;
  if (i < N) deg[i] = 1.0f;  // self loop
}

__global__ void k_deg_edges(const int* __restrict__ dst, float* __restrict__ deg, int E) {
  int i = blockIdx.x * blockDim.x + threadIdx.x;
  if (i < E) atomicAdd(&deg[dst[i]], 1.0f);
}

__global__ void k_dinv(const float* __restrict__ deg, float* __restrict__ dinv, int N) {
  int i = blockIdx.x * blockDim.x + threadIdx.x;
  if (i < N) dinv[i] = rsqrtf(deg[i]);  // deg >= 1 always
}

// ---------------- bf16 WMMA GEMM: C[M x NOUT] = A[M x 128] * Bt[NOUT x 128]^T ----------------
template <int NOUT>
__global__ __launch_bounds__(256)
void k_gemm_bf16(const unsigned short* __restrict__ A,
                 const unsigned short* __restrict__ Bt,
                 float* __restrict__ C, int M) {
  constexpr int K  = 128;
  constexpr int TN = NOUT / 16;
  int wave = blockIdx.x * (blockDim.x >> 5) + (threadIdx.x >> 5);
  int lane = threadIdx.x & 31;
  int tilesM = M >> 4;
  if (wave >= tilesM * TN) return;
  int tm = wave / TN, tn = wave - tm * TN;
  int half = lane >> 4;       // lane-group selects K sub-blocks
  int l15  = lane & 15;       // row (A) / col (B)

  const char* Arow = (const char*)(A + (size_t)(tm * 16 + l15) * K);
  const char* Brow = (const char*)(Bt + (size_t)(tn * 16 + l15) * K);

  v8f acc = {};
#pragma unroll
  for (int kb = 0; kb < K / 32; ++kb) {
    int kbase = kb * 32 + half * 8;
    FragBF fa, fb;
    fa.q[0] = *(const u32x4*)(Arow + (size_t)kbase * 2);
    fa.q[1] = *(const u32x4*)(Arow + (size_t)(kbase + 16) * 2);
    fb.q[0] = *(const u32x4*)(Brow + (size_t)kbase * 2);
    fb.q[1] = *(const u32x4*)(Brow + (size_t)(kbase + 16) * 2);
    acc = __builtin_amdgcn_wmma_f32_16x16x32_bf16(false, fa.v, false, fb.v,
                                                  (short)0, acc, false, false);
  }
  // C/D layout: VGPR r -> row 16*tm + 8*half + r, col 16*tn + l15
  float* Cout = C + (size_t)(tm * 16 + half * 8) * NOUT + tn * 16 + l15;
#pragma unroll
  for (int r = 0; r < 8; ++r) Cout[(size_t)r * NOUT] = acc[r];
}

// ---------------- self-loop init: agg = hw * dinv[n]^2 (full overwrite) ----------------
__global__ void k_selfloop(const float* __restrict__ hw, const float* __restrict__ dinv,
                           float* __restrict__ agg, int N, int C4) {  // C4 = channels/4
  int i = blockIdx.x * blockDim.x + threadIdx.x;
  if (i >= N * C4) return;
  int n = i / C4;
  float di = dinv[n];
  float s = di * di;
  float4 v = ((const float4*)hw)[i];
  float4 o;
  o.x = v.x * s; o.y = v.y * s; o.z = v.z * s; o.w = v.w * s;
  ((float4*)agg)[i] = o;
}

// ---------------- edge scatter (wave32 per edge) ----------------
__global__ void k_scatter128(const float* __restrict__ hw, const int* __restrict__ src,
                             const int* __restrict__ dst, const float* __restrict__ dinv,
                             float* __restrict__ agg, int E) {
  int gid = blockIdx.x * blockDim.x + threadIdx.x;
  int e = gid >> 5, lane = gid & 31;
  if (e >= E) return;
  int s = src[e], d = dst[e];
  float nrm = dinv[s] * dinv[d];
  float4 v = ((const float4*)(hw + (size_t)s * 128))[lane];  // coalesced 512B gather
  float* o = agg + (size_t)d * 128 + lane * 4;
  atomicAdd(o + 0, v.x * nrm);
  atomicAdd(o + 1, v.y * nrm);
  atomicAdd(o + 2, v.z * nrm);
  atomicAdd(o + 3, v.w * nrm);
}

__global__ void k_scatter64(const float* __restrict__ hw, const int* __restrict__ src,
                            const int* __restrict__ dst, const float* __restrict__ dinv,
                            float* __restrict__ agg, int E) {
  int gid = blockIdx.x * blockDim.x + threadIdx.x;
  int e = gid >> 5, lane = gid & 31;
  if (e >= E) return;
  int s = src[e], d = dst[e];
  float nrm = dinv[s] * dinv[d];
  float2 v = ((const float2*)(hw + (size_t)s * 64))[lane];
  float* o = agg + (size_t)d * 64 + lane * 2;
  atomicAdd(o + 0, v.x * nrm);
  atomicAdd(o + 1, v.y * nrm);
}

// ---------------- bias + relu + cast to bf16 ----------------
__global__ void k_bias_relu_bf16(const float* __restrict__ agg, const float* __restrict__ b,
                                 unsigned short* __restrict__ out, int total, int Cmask) {
  int i = blockIdx.x * blockDim.x + threadIdx.x;
  if (i >= total) return;
  out[i] = f2bf(fmaxf(agg[i] + b[i & Cmask], 0.0f));
}

// ---------------- final: relu(agg+b1) dot Wl + bl -> [-logit, logit] ----------------
__global__ void k_final(const float* __restrict__ agg, const float* __restrict__ b1,
                        const float* __restrict__ Wl, const float* __restrict__ bl,
                        float* __restrict__ out, int N) {
  int gid = blockIdx.x * blockDim.x + threadIdx.x;
  int n = gid >> 5, lane = gid & 31;
  if (n >= N) return;
  float2 h  = ((const float2*)(agg + (size_t)n * 64))[lane];
  float2 bb = ((const float2*)b1)[lane];
  float2 w  = ((const float2*)Wl)[lane];
  float p = fmaxf(h.x + bb.x, 0.0f) * w.x + fmaxf(h.y + bb.y, 0.0f) * w.y;
#pragma unroll
  for (int o = 16; o > 0; o >>= 1) p += __shfl_xor(p, o, 32);
  if (lane == 0) {
    float logit = p + bl[0];
    out[2 * n + 0] = -logit;
    out[2 * n + 1] = logit;
  }
}

extern "C" void kernel_launch(void* const* d_in, const int* in_sizes, int n_in,
                              void* d_out, int out_size, void* d_ws, size_t ws_size,
                              hipStream_t stream) {
  const float* x  = (const float*)d_in[0];
  const int*   ei = (const int*)d_in[1];
  const float* W0 = (const float*)d_in[2];
  const float* b0 = (const float*)d_in[3];
  const float* W1 = (const float*)d_in[4];
  const float* b1 = (const float*)d_in[5];
  const float* Wl = (const float*)d_in[6];
  const float* bl = (const float*)d_in[7];

  const int IN = 128, H = 128, H2 = 64;
  int N = in_sizes[0] / IN;
  int E = in_sizes[1] / 2;
  const int* src = ei;
  const int* dst = ei + E;

  // ---- workspace carving (all 256B aligned) ----
  char* ws = (char*)d_ws;
  size_t off = 0;
  auto carve = [&](size_t bytes) -> char* {
    char* p = ws + off;
    off = (off + bytes + 255) & ~(size_t)255;
    return p;
  };
  float*          deg  = (float*)carve((size_t)N * 4);
  float*          dinv = (float*)carve((size_t)N * 4);
  unsigned short* Xb   = (unsigned short*)carve((size_t)N * IN * 2);
  unsigned short* W0b  = (unsigned short*)carve((size_t)IN * H * 2);
  unsigned short* W1b  = (unsigned short*)carve((size_t)H * H2 * 2);
  float*          HW0  = (float*)carve((size_t)N * H * 4);
  float*          AGG0 = (float*)carve((size_t)N * H * 4);
  // reuse (strictly sequential stream ordering makes this safe):
  unsigned short* H1b  = Xb;                       // Xb dead after GEMM0
  float*          HW1  = HW0;                      // HW0 dead after scatter128
  float*          AGG1 = HW0 + (size_t)N * H2;     // second half of HW0 block

  const int T = 256;
  auto blk = [](long long n, int t) { return (unsigned)((n + t - 1) / t); };

  // degree / normalization
  k_deg_init<<<blk(N, T), T, 0, stream>>>(deg, N);
  k_deg_edges<<<blk(E, T), T, 0, stream>>>(dst, deg, E);
  k_dinv<<<blk(N, T), T, 0, stream>>>(deg, dinv, N);

  // bf16 operand prep
  k_cast_bf16<<<blk((long long)N * IN, T), T, 0, stream>>>(x, Xb, N * IN);
  k_transpose_bf16<<<blk(IN * H, T), T, 0, stream>>>(W0, W0b, IN, H);
  k_transpose_bf16<<<blk(H * H2, T), T, 0, stream>>>(W1, W1b, H, H2);

  // layer 0: GEMM -> self-loop -> edge scatter -> bias+relu+cast
  long long waves0 = (long long)(N / 16) * (H / 16);
  k_gemm_bf16<128><<<blk(waves0 * 32, T), T, 0, stream>>>(Xb, W0b, HW0, N);
  k_selfloop<<<blk((long long)N * (H / 4), T), T, 0, stream>>>(HW0, dinv, AGG0, N, H / 4);
  k_scatter128<<<blk((long long)E * 32, T), T, 0, stream>>>(HW0, src, dst, dinv, AGG0, E);
  k_bias_relu_bf16<<<blk((long long)N * H, T), T, 0, stream>>>(AGG0, b0, H1b, N * H, H - 1);

  // layer 1
  long long waves1 = (long long)(N / 16) * (H2 / 16);
  k_gemm_bf16<64><<<blk(waves1 * 32, T), T, 0, stream>>>(H1b, W1b, HW1, N);
  k_selfloop<<<blk((long long)N * (H2 / 4), T), T, 0, stream>>>(HW1, dinv, AGG1, N, H2 / 4);
  k_scatter64<<<blk((long long)E * 32, T), T, 0, stream>>>(HW1, src, dst, dinv, AGG1, E);

  // head
  k_final<<<blk((long long)N * 32, T), T, 0, stream>>>(AGG1, b1, Wl, bl, (float*)d_out, N);
}
